// Attention_38843684225791
// MI455X (gfx1250) — matrix operand order repaired and checked by reading, not compiled, not run on hardware
//
#include <hip/hip_runtime.h>
#include <hip/hip_bf16.h>
#include <cstdint>

// ---------- problem constants ----------
#define BB 4
#define SS 2048
#define CC 768
#define HH 12
#define HD 64
#define MASK_NEG -100000.0f

typedef __bf16 bf16_t;
typedef __attribute__((ext_vector_type(8)))  __bf16 v8bf;
typedef __attribute__((ext_vector_type(16))) __bf16 v16bf;
typedef __attribute__((ext_vector_type(8)))  float  v8f;

static __device__ __forceinline__ v16bf cat16(v8bf a, v8bf b) {
    return __builtin_shufflevector(a, b, 0,1,2,3,4,5,6,7,8,9,10,11,12,13,14,15);
}
// 16 contiguous bf16 (B-matrix per-lane pattern: lane holds one column's 16 K values)
static __device__ __forceinline__ v16bf ld16(const bf16_t* p) {
    v8bf a = *(const v8bf*)p;
    v8bf b = *(const v8bf*)(p + 8);
    return cat16(a, b);
}
// A-matrix 16-bit 16x32 per-lane pattern (contiguous K, matches ld16 B):
//  lanes 0-15 : K {0..15};  lanes 16-31 : K {16..31}
static __device__ __forceinline__ v16bf ldA(const bf16_t* row, int lh) {
    v8bf a = *(const v8bf*)(row + 16 * lh);
    v8bf b = *(const v8bf*)(row + 16 * lh + 8);
    return cat16(a, b);
}
static __device__ __forceinline__ v8f wmma_bf16(v16bf a, v16bf b, v8f c) {
    return __builtin_amdgcn_wmma_f32_16x16x32_bf16(false, a, false, b, (short)0, c, false, false);
}
// gfx1250 async copy global->LDS (16B), tracked with ASYNCcnt
static __device__ __forceinline__ void async_cp16(void* lds_ptr, const void* gptr) {
    uint32_t lds_off = (uint32_t)(uintptr_t)lds_ptr;   // LDS aperture: addr[31:0] = LDS offset
    asm volatile("global_load_async_to_lds_b128 %0, %1, off"
                 :: "v"(lds_off), "v"((uint64_t)(uintptr_t)gptr)
                 : "memory");
}
static __device__ __forceinline__ void wait_async0() {
    asm volatile("s_wait_asynccnt 0x0" ::: "memory");
}

// =====================================================================
// Kernel 1: QKV projection. x[8192,768] @ w_qkv[2304,768]^T
// Tile: BM=128, BN=64, BK=32. 8 waves: 4(M) x 2(N), each wave 32x32.
// BN=64 == one head's width -> (three, head) are blockIdx.y-uniform (SGPR).
// Scatter: q[b,h,s,d] (pre-scaled 1/8), k[b,h,s,d], vT[b,h,d,s]  (bf16)
// =====================================================================
__global__ __launch_bounds__(256) void qkv_gemm_kernel(
    const float* __restrict__ x, const float* __restrict__ wqkv,
    bf16_t* __restrict__ qw, bf16_t* __restrict__ kw, bf16_t* __restrict__ vw)
{
    __shared__ __align__(16) bf16_t As[128][40];
    __shared__ __align__(16) bf16_t Bs[64][40];

    const int tid  = threadIdx.x;
    const int w    = tid >> 5;
    const int lane = tid & 31;
    const int r    = lane & 15;
    const int lh   = lane >> 4;
    const int wm   = w & 3;      // M quadrant (32 rows each)
    const int wn   = w >> 2;     // N half (32 cols each)
    const int m0   = blockIdx.x * 128;
    const int n0   = blockIdx.y * 64;

    v8f acc[2][2];
    const v8f zacc = {0.f,0.f,0.f,0.f,0.f,0.f,0.f,0.f};
#pragma unroll
    for (int i = 0; i < 2; ++i)
#pragma unroll
        for (int j = 0; j < 2; ++j) acc[i][j] = zacc;

    for (int kb = 0; kb < CC / 32; ++kb) {
        __syncthreads();
        // stage A: 128x32 fp32 -> bf16 (batch loads, then convert+store)
        float4 ta[4];
#pragma unroll
        for (int i = 0; i < 4; ++i) {
            int e = tid + i * 256, row = e >> 3, cv = e & 7;
            ta[i] = *(const float4*)(x + (size_t)(m0 + row) * CC + kb * 32 + cv * 4);
        }
        // stage B: 64x32 fp32 -> bf16
        float4 tb[2];
#pragma unroll
        for (int i = 0; i < 2; ++i) {
            int e = tid + i * 256, row = e >> 3, cv = e & 7;
            tb[i] = *(const float4*)(wqkv + (size_t)(n0 + row) * CC + kb * 32 + cv * 4);
        }
        __builtin_prefetch(x + (size_t)(m0 + (tid >> 3)) * CC + (kb + 1) * 32, 0, 1);
#pragma unroll
        for (int i = 0; i < 4; ++i) {
            int e = tid + i * 256, row = e >> 3, cv = e & 7;
            bf16_t* d = &As[row][cv * 4];
            d[0] = (bf16_t)ta[i].x; d[1] = (bf16_t)ta[i].y;
            d[2] = (bf16_t)ta[i].z; d[3] = (bf16_t)ta[i].w;
        }
#pragma unroll
        for (int i = 0; i < 2; ++i) {
            int e = tid + i * 256, row = e >> 3, cv = e & 7;
            bf16_t* d = &Bs[row][cv * 4];
            d[0] = (bf16_t)tb[i].x; d[1] = (bf16_t)tb[i].y;
            d[2] = (bf16_t)tb[i].z; d[3] = (bf16_t)tb[i].w;
        }
        __syncthreads();

        v16bf af[2], bfr[2];
#pragma unroll
        for (int ti = 0; ti < 2; ++ti)
            af[ti] = ldA(&As[wm * 32 + ti * 16 + r][0], lh);
#pragma unroll
        for (int tj = 0; tj < 2; ++tj)
            bfr[tj] = ld16(&Bs[wn * 32 + tj * 16 + r][16 * lh]);
#pragma unroll
        for (int ti = 0; ti < 2; ++ti)
#pragma unroll
            for (int tj = 0; tj < 2; ++tj)
                acc[ti][tj] = wmma_bf16(af[ti], bfr[tj], acc[ti][tj]);
    }

    // ---- scatter epilogue: (three, head) uniform per block; (b, s-base) too
    const int three = blockIdx.y / HH;             // 0,1,2        (scalar)
    const int hh    = blockIdx.y - three * HH;     // head         (scalar)
    const int bb    = blockIdx.x >> 4;             // batch        (scalar)
    const int sbase = (blockIdx.x & 15) * 128;     // row in seq   (scalar)
    const size_t bhOff = (size_t)(bb * HH + hh);
#pragma unroll
    for (int ti = 0; ti < 2; ++ti)
#pragma unroll
        for (int tj = 0; tj < 2; ++tj)
#pragma unroll
            for (int v = 0; v < 8; ++v) {
                int s = sbase + wm * 32 + ti * 16 + v + 8 * lh;
                int d = wn * 32 + tj * 16 + r;     // 0..63 within head
                float val = acc[ti][tj][v];
                if (three == 0)
                    qw[(bhOff * SS + s) * HD + d] = (bf16_t)(val * 0.125f);
                else if (three == 1)
                    kw[(bhOff * SS + s) * HD + d] = (bf16_t)val;
                else
                    vw[(bhOff * HD + d) * SS + s] = (bf16_t)val;
            }
}

// =====================================================================
// Kernel 2: flash attention. grid = (S/128, H, B), 8 waves/WG,
// each wave owns 16 query rows; online softmax over 64 key tiles of 32.
// K/V tiles staged into LDS once per WG via async-to-LDS, double-buffered.
// =====================================================================
__global__ __launch_bounds__(256) void flash_attn_kernel(
    const bf16_t* __restrict__ qw, const bf16_t* __restrict__ kw,
    const bf16_t* __restrict__ vw, const int* __restrict__ mask,
    bf16_t* __restrict__ ao)
{
    __shared__ __align__(16) bf16_t Ks[2][32][64];    // [buf][key][d]
    __shared__ __align__(16) bf16_t Vs[2][64][32];    // [buf][d][key]
    __shared__ __align__(16) bf16_t plds[8][16][40];  // per-wave P re-swizzle tile

    const int tid  = threadIdx.x;
    const int w    = tid >> 5;
    const int lane = tid & 31;
    const int r    = lane & 15;
    const int lh   = lane >> 4;
    const int b    = blockIdx.z;
    const int h    = blockIdx.y;
    const int q0   = blockIdx.x * 128 + w * 16;

    const size_t bh   = (size_t)b * HH + h;
    const bf16_t* qbase = qw + bh * SS * HD;
    const bf16_t* kbase = kw + bh * SS * HD;
    const bf16_t* vbase = vw + bh * HD * SS;
    const int*    mbase = mask + (size_t)b * SS * SS;

    // async-stage one 32-key K/V tile into LDS buffer `buf`
    const int krow = tid >> 3, kc8 = (tid & 7) * 8;   // 256 x 16B for K tile
    const int vrow = tid >> 2, vc8 = (tid & 3) * 8;   // 256 x 16B for V tile
    auto issue_tile = [&](int kt, int buf) {
        async_cp16(&Ks[buf][krow][kc8], kbase + (size_t)(kt * 32 + krow) * HD + kc8);
        async_cp16(&Vs[buf][vrow][vc8], vbase + (size_t)vrow * SS + kt * 32 + vc8);
    };

    // Q as two A-fragments (K = 0..31, 32..63), pre-scaled at store time
    v16bf qa[2];
#pragma unroll
    for (int kk = 0; kk < 2; ++kk)
        qa[kk] = ldA(qbase + (size_t)(q0 + r) * HD + kk * 32, lh);

    const v8f zacc = {0.f,0.f,0.f,0.f,0.f,0.f,0.f,0.f};
    v8f o[4];
#pragma unroll
    for (int dt = 0; dt < 4; ++dt) o[dt] = zacc;
    float mrow[8], lrow[8];
#pragma unroll
    for (int v = 0; v < 8; ++v) { mrow[v] = -3.0e38f; lrow[v] = 0.0f; }

    issue_tile(0, 0);  // prologue

    for (int kt = 0; kt < SS / 32; ++kt) {
        const int buf = kt & 1;
        wait_async0();          // this wave's tile loads complete
        __syncthreads();        // all waves' tiles visible in LDS
        if (kt + 1 < SS / 32) issue_tile(kt + 1, buf ^ 1);

        // ---- S = Q K^T (two 16x16 N-tiles, K=64 split as 2 WMMAs each)
        v8f sc[2];
#pragma unroll
        for (int nt = 0; nt < 2; ++nt) {
            const bf16_t* kr = &Ks[buf][nt * 16 + r][0];
            v8f s = zacc;
            s = wmma_bf16(qa[0], ld16(kr + 0  + 16 * lh), s);
            s = wmma_bf16(qa[1], ld16(kr + 32 + 16 * lh), s);
            sc[nt] = s;
        }

        // ---- mask bias + online softmax stats
        float mnew[8], alpha[8];
#pragma unroll
        for (int v = 0; v < 8; ++v) {
            int qrow = q0 + v + 8 * lh;
            const int* mr = mbase + (size_t)qrow * SS + kt * 32 + r;
            float s0 = sc[0][v] + (float)mr[0]  * MASK_NEG;
            float s1 = sc[1][v] + (float)mr[16] * MASK_NEG;
            sc[0][v] = s0; sc[1][v] = s1;
            float mx = fmaxf(s0, s1);
#pragma unroll
            for (int off = 1; off < 16; off <<= 1)
                mx = fmaxf(mx, __shfl_xor(mx, off, 32));
            mnew[v]  = fmaxf(mrow[v], mx);
            alpha[v] = __expf(mrow[v] - mnew[v]);
            mrow[v]  = mnew[v];
        }

        // ---- P = exp(S - m), row sums, rescale O, write P to LDS tile
#pragma unroll
        for (int v = 0; v < 8; ++v) {
            float p0 = __expf(sc[0][v] - mnew[v]);
            float p1 = __expf(sc[1][v] - mnew[v]);
            float rs = p0 + p1;
#pragma unroll
            for (int off = 1; off < 16; off <<= 1)
                rs += __shfl_xor(rs, off, 32);
            lrow[v] = lrow[v] * alpha[v] + rs;
            int prow = v + 8 * lh;
            plds[w][prow][r]      = (bf16_t)p0;
            plds[w][prow][16 + r] = (bf16_t)p1;
#pragma unroll
            for (int dt = 0; dt < 4; ++dt) o[dt][v] *= alpha[v];
        }

        __syncthreads();
        // re-read P in A-matrix layout (16x32, K = 32 keys)
        v16bf pa = ldA(&plds[w][r][0], lh);

        // ---- O += P V  (Vs gives contiguous K along key index)
#pragma unroll
        for (int dt = 0; dt < 4; ++dt) {
            const bf16_t* vr = &Vs[buf][dt * 16 + r][16 * lh];
            o[dt] = wmma_bf16(pa, ld16(vr), o[dt]);
        }
        __syncthreads();  // protect plds (and Ks/Vs[buf]) before next iter's writes
    }

    // ---- normalize + store bf16 [B,S,C] with head-concat
#pragma unroll
    for (int v = 0; v < 8; ++v) {
        float inv = 1.0f / lrow[v];
        int srow = q0 + v + 8 * lh;
#pragma unroll
        for (int dt = 0; dt < 4; ++dt) {
            int c = h * HD + dt * 16 + r;
            ao[((size_t)b * SS + srow) * CC + c] = (bf16_t)(o[dt][v] * inv);
        }
    }
}

// =====================================================================
// Kernel 3: output projection. ao[8192,768](bf16) @ w_proj[768,768]^T + b
// A tile staged with async-to-LDS (already bf16); fp32 output.
// =====================================================================
__global__ __launch_bounds__(256) void proj_gemm_kernel(
    const bf16_t* __restrict__ a, const float* __restrict__ wproj,
    const float* __restrict__ bias, float* __restrict__ out)
{
    __shared__ __align__(16) bf16_t As[128][40];
    __shared__ __align__(16) bf16_t Bs[64][40];

    const int tid  = threadIdx.x;
    const int w    = tid >> 5;
    const int lane = tid & 31;
    const int r    = lane & 15;
    const int lh   = lane >> 4;
    const int wm   = w & 3;
    const int wn   = w >> 2;
    const int m0   = blockIdx.x * 128;
    const int n0   = blockIdx.y * 64;

    v8f acc[2][2];
    const v8f zacc = {0.f,0.f,0.f,0.f,0.f,0.f,0.f,0.f};
#pragma unroll
    for (int i = 0; i < 2; ++i)
#pragma unroll
        for (int j = 0; j < 2; ++j) acc[i][j] = zacc;

    for (int kb = 0; kb < CC / 32; ++kb) {
        __syncthreads();
        // stage A: 128x32 bf16 via async global->LDS (512 x 16B)
#pragma unroll
        for (int i = 0; i < 2; ++i) {
            int e = tid + i * 256, row = e >> 2, cv = e & 3;
            async_cp16(&As[row][cv * 8],
                       a + (size_t)(m0 + row) * CC + kb * 32 + cv * 8);
        }
        // stage B: 64x32 fp32 -> bf16 (batched)
        float4 tb[2];
#pragma unroll
        for (int i = 0; i < 2; ++i) {
            int e = tid + i * 256, row = e >> 3, cv = e & 7;
            tb[i] = *(const float4*)(wproj + (size_t)(n0 + row) * CC + kb * 32 + cv * 4);
        }
        __builtin_prefetch(wproj + (size_t)(n0 + ((tid >> 3) & 63)) * CC + (kb + 1) * 32, 0, 1);
#pragma unroll
        for (int i = 0; i < 2; ++i) {
            int e = tid + i * 256, row = e >> 3, cv = e & 7;
            bf16_t* d = &Bs[row][cv * 4];
            d[0] = (bf16_t)tb[i].x; d[1] = (bf16_t)tb[i].y;
            d[2] = (bf16_t)tb[i].z; d[3] = (bf16_t)tb[i].w;
        }
        wait_async0();
        __syncthreads();

        v16bf af[2], bfr[2];
#pragma unroll
        for (int ti = 0; ti < 2; ++ti)
            af[ti] = ldA(&As[wm * 32 + ti * 16 + r][0], lh);
#pragma unroll
        for (int tj = 0; tj < 2; ++tj)
            bfr[tj] = ld16(&Bs[wn * 32 + tj * 16 + r][16 * lh]);
#pragma unroll
        for (int ti = 0; ti < 2; ++ti)
#pragma unroll
            for (int tj = 0; tj < 2; ++tj)
                acc[ti][tj] = wmma_bf16(af[ti], bfr[tj], acc[ti][tj]);
    }

#pragma unroll
    for (int ti = 0; ti < 2; ++ti)
#pragma unroll
        for (int tj = 0; tj < 2; ++tj)
#pragma unroll
            for (int v = 0; v < 8; ++v) {
                int m = m0 + wm * 32 + ti * 16 + v + 8 * lh;
                int n = n0 + wn * 32 + tj * 16 + r;
                out[(size_t)m * CC + n] = acc[ti][tj][v] + bias[n];
            }
}

// =====================================================================
extern "C" void kernel_launch(void* const* d_in, const int* in_sizes, int n_in,
                              void* d_out, int out_size, void* d_ws, size_t ws_size,
                              hipStream_t stream) {
    const float* x     = (const float*)d_in[0];
    const int*   mask  = (const int*)d_in[1];
    const float* wqkv  = (const float*)d_in[2];
    const float* wproj = (const float*)d_in[3];
    const float* bproj = (const float*)d_in[4];
    float* out = (float*)d_out;

    // workspace layout (bf16): q, k, vT, attn_out — each B*H*S*HD = 6,291,456 elems
    const size_t seg = (size_t)BB * HH * SS * HD * sizeof(bf16_t); // 12,582,912 B
    char* ws = (char*)d_ws;
    bf16_t* qw = (bf16_t*)(ws + 0 * seg);
    bf16_t* kw = (bf16_t*)(ws + 1 * seg);
    bf16_t* vw = (bf16_t*)(ws + 2 * seg);
    bf16_t* ao = (bf16_t*)(ws + 3 * seg);

    // 1) QKV projection:  M=8192 / BM=128 -> 64,  N=2304 / BN=64 -> 36
    qkv_gemm_kernel<<<dim3(64, 36), 256, 0, stream>>>(x, wqkv, qw, kw, vw);
    // 2) flash attention: (S/128=16, H=12, B=4)
    flash_attn_kernel<<<dim3(16, HH, BB), 256, 0, stream>>>(qw, kw, vw, mask, ao);
    // 3) output projection: M=8192/128 -> 64, N=768/64 -> 12
    proj_gemm_kernel<<<dim3(64, 12), 256, 0, stream>>>(ao, wproj, bproj, out);
}
